// Matmul_27066883899829
// MI455X (gfx1250) — 8-bit matrix operand order repaired and checked by reading, compiled, not run on hardware
//
#include <hip/hip_runtime.h>
#include <hip/hip_bf16.h>
#include <stdint.h>

#define M_DIM 4096
#define N_DIM 8192
#define K_DIM 2048
#define CLIP 127.0f

typedef int v8i __attribute__((ext_vector_type(8)));

__global__ void init_scales_kernel(unsigned* __restrict__ s) {
    if (threadIdx.x < 2) s[threadIdx.x] = 0u;
}

__global__ void reduce_maxabs_kernel(const float* __restrict__ x, size_t n4,
                                     unsigned* __restrict__ target) {
    __shared__ float red[256];
    const float4* x4 = (const float4*)x;
    float m = 0.0f;
    for (size_t i = (size_t)blockIdx.x * blockDim.x + threadIdx.x; i < n4;
         i += (size_t)gridDim.x * blockDim.x) {
        float4 v = x4[i];
        m = fmaxf(m, fabsf(v.x));
        m = fmaxf(m, fabsf(v.y));
        m = fmaxf(m, fabsf(v.z));
        m = fmaxf(m, fabsf(v.w));
    }
    red[threadIdx.x] = m;
    __syncthreads();
    for (int s = 128; s > 0; s >>= 1) {
        if ((int)threadIdx.x < s)
            red[threadIdx.x] = fmaxf(red[threadIdx.x], red[threadIdx.x + s]);
        __syncthreads();
    }
    if (threadIdx.x == 0) atomicMax(target, __float_as_uint(red[0]));
}

__device__ __forceinline__ float load_scale(const unsigned* mb) {
    return CLIP / fmaxf(__uint_as_float(*mb), 1e-6f);
}

__device__ __forceinline__ signed char quant1(float v, float scale) {
    return (signed char)(int)fminf(fmaxf(rintf(v * scale), -CLIP), CLIP);
}

__global__ void quant_rowmajor_kernel(const float* __restrict__ x,
                                      signed char* __restrict__ q, size_t n4,
                                      const unsigned* __restrict__ mb) {
    float scale = load_scale(mb);
    const float4* x4 = (const float4*)x;
    char4* q4 = (char4*)q;
    for (size_t i = (size_t)blockIdx.x * blockDim.x + threadIdx.x; i < n4;
         i += (size_t)gridDim.x * blockDim.x) {
        float4 v = x4[i];
        char4 c;
        c.x = quant1(v.x, scale);
        c.y = quant1(v.y, scale);
        c.z = quant1(v.z, scale);
        c.w = quant1(v.w, scale);
        q4[i] = c;
    }
}

__global__ void quant_transpose_kernel(const float* __restrict__ x,
                                       signed char* __restrict__ qT,
                                       const unsigned* __restrict__ mb) {
    __shared__ signed char tile[32][33];
    float scale = load_scale(mb);
    int n0 = blockIdx.x * 32;
    int k0 = blockIdx.y * 32;
    int tx = threadIdx.x;
    int ty = threadIdx.y;
#pragma unroll
    for (int i = 0; i < 4; ++i) {
        int k = k0 + ty + i * 8;
        tile[ty + i * 8][tx] = quant1(x[(size_t)k * N_DIM + n0 + tx], scale);
    }
    __syncthreads();
#pragma unroll
    for (int i = 0; i < 4; ++i) {
        int nn = n0 + ty + i * 8;
        qT[(size_t)nn * K_DIM + k0 + tx] = tile[tx][ty + i * 8];
    }
}

__global__ void __launch_bounds__(256)
gemm_i8_wmma_kernel(const signed char* __restrict__ Aq,
                    const signed char* __restrict__ BqT,
                    float* __restrict__ out,
                    const unsigned* __restrict__ scales) {
    const int lane   = threadIdx.x & 31;
    const int wave   = threadIdx.x >> 5;
    const int waveM  = wave >> 2;
    const int waveN  = wave & 3;
    const int laneLo = lane & 15;
    const int half   = lane >> 4;

    const int mBase = blockIdx.y * 128 + waveM * 64;
    const int nBase = blockIdx.x * 128 + waveN * 32;

    v8i acc[4][2];
#pragma unroll
    for (int t = 0; t < 4; ++t)
#pragma unroll
        for (int u = 0; u < 2; ++u) {
            v8i z = {};
            acc[t][u] = z;
        }

    const signed char* aPtr[4];
#pragma unroll
    for (int t = 0; t < 4; ++t)
        aPtr[t] = Aq + (size_t)(mBase + t * 16 + laneLo) * K_DIM + half * 16;
    const signed char* bPtr[2];
#pragma unroll
    for (int u = 0; u < 2; ++u)
        bPtr[u] = BqT + (size_t)(nBase + u * 16 + laneLo) * K_DIM + half * 16;

    for (int k0 = 0; k0 < K_DIM; k0 += 64) {
        v8i afrag[4];
#pragma unroll
        for (int t = 0; t < 4; ++t) {
            const uint2* p = (const uint2*)(aPtr[t] + k0);
            uint2 d0 = p[0];
            uint2 d1 = p[1];
            uint2 d2 = p[4];
            uint2 d3 = p[5];
            v8i a;
            a[0] = (int)d0.x; a[1] = (int)d0.y;
            a[2] = (int)d1.x; a[3] = (int)d1.y;
            a[4] = (int)d2.x; a[5] = (int)d2.y;
            a[6] = (int)d3.x; a[7] = (int)d3.y;
            afrag[t] = a;
        }
        v8i bfrag[2];
#pragma unroll
        for (int u = 0; u < 2; ++u) {
            const uint4* p = (const uint4*)(bPtr[u] + k0);
            uint4 q0 = p[0];
            uint4 q1 = p[2];
            v8i b;
            b[0] = (int)q0.x; b[1] = (int)q0.y; b[2] = (int)q0.z; b[3] = (int)q0.w;
            b[4] = (int)q1.x; b[5] = (int)q1.y; b[6] = (int)q1.z; b[7] = (int)q1.w;
            bfrag[u] = b;
        }
#pragma unroll
        for (int t = 0; t < 4; ++t)
#pragma unroll
            for (int u = 0; u < 2; ++u)
                acc[t][u] = __builtin_amdgcn_wmma_i32_16x16x64_iu8(
                     true, afrag[t],  true, bfrag[u],
                    acc[t][u],  false,  false);
    }

    float sL = load_scale(scales + 0);
    float sR = load_scale(scales + 1);
    float inv = 1.0f / (sL * sR);

#pragma unroll
    for (int t = 0; t < 4; ++t) {
#pragma unroll
        for (int u = 0; u < 2; ++u) {
            int row0 = mBase + t * 16 + half * 8;
            int col  = nBase + u * 16 + laneLo;
#pragma unroll
            for (int j = 0; j < 8; ++j)
                out[(size_t)(row0 + j) * N_DIM + col] =
                    (float)acc[t][u][j] * inv;
        }
    }
}

extern "C" void kernel_launch(void* const* d_in, const int* in_sizes, int n_in,
                              void* d_out, int out_size, void* d_ws, size_t ws_size,
                              hipStream_t stream) {
    const float* lhs = (const float*)d_in[0];
    const float* rhs = (const float*)d_in[1];
    float* out = (float*)d_out;

    unsigned* scales   = (unsigned*)d_ws;
    signed char* lhs_q = (signed char*)d_ws + 256;
    signed char* rhs_qT =
        (signed char*)d_ws + 256 + (size_t)M_DIM * K_DIM;

    init_scales_kernel<<<1, 32, 0, stream>>>(scales);

    reduce_maxabs_kernel<<<2048, 256, 0, stream>>>(
        lhs, (size_t)M_DIM * K_DIM / 4, scales + 0);
    reduce_maxabs_kernel<<<2048, 256, 0, stream>>>(
        rhs, (size_t)K_DIM * N_DIM / 4, scales + 1);

    quant_rowmajor_kernel<<<2048, 256, 0, stream>>>(
        lhs, lhs_q, (size_t)M_DIM * K_DIM / 4, scales + 0);

    dim3 tgrid(N_DIM / 32, K_DIM / 32);
    quant_transpose_kernel<<<tgrid, dim3(32, 8), 0, stream>>>(
        rhs, rhs_qT, scales + 1);

    dim3 ggrid(N_DIM / 128, M_DIM / 128);
    gemm_i8_wmma_kernel<<<ggrid, 256, 0, stream>>>(lhs_q, rhs_qT, out, scales);
}
